// Net_34273839022236
// MI455X (gfx1250) — compile-verified, matrix operand order checked by reading, not run on hardware
//
#include <hip/hip_runtime.h>
#include <math.h>

typedef __attribute__((ext_vector_type(16))) __bf16 v16bf;
typedef __attribute__((ext_vector_type(8)))  __bf16 v8bf;
typedef __attribute__((ext_vector_type(8)))  float  v8f;

#define GK 256   // inner (K) dimension of every GEMM in this net

// ---------------- utility kernels ----------------

__global__ void fill_f32(float* __restrict__ p, float v, size_t n) {
    size_t t = (size_t)blockIdx.x * blockDim.x + threadIdx.x;
    if (t < n) p[t] = v;
}

__global__ void deg_accum(const int* __restrict__ dst, const float* __restrict__ w,
                          float* __restrict__ deg, int E) {
    int e = blockIdx.x * blockDim.x + threadIdx.x;
    if (e < E) atomicAdd(&deg[dst[e]], w[e]);
}

__global__ void dinv_kernel(const float* __restrict__ deg, float* __restrict__ dinv, int N) {
    int i = blockIdx.x * blockDim.x + threadIdx.x;
    if (i < N) { float d = deg[i]; dinv[i] = d > 0.f ? rsqrtf(d) : 0.f; }
}

__global__ void norm_kernel(const int* __restrict__ src, const int* __restrict__ dst,
                            const float* __restrict__ w, const float* __restrict__ dinv,
                            float* __restrict__ nrm, int E) {
    int e = blockIdx.x * blockDim.x + threadIdx.x;
    if (e < E) nrm[e] = dinv[src[e]] * w[e] * dinv[dst[e]];
}

__global__ void cvt_bf16(const float* __restrict__ x, __bf16* __restrict__ y, size_t n) {
    size_t t = (size_t)blockIdx.x * blockDim.x + threadIdx.x;
    if (t < n) y[t] = (__bf16)x[t];
}

// W: [GK, F] row-major fp32  ->  Wt: [Fpad, GK] row-major bf16 (zero-padded rows)
__global__ void cvt_wt(const float* __restrict__ W, __bf16* __restrict__ Wt, int F, int Fpad) {
    int t = blockIdx.x * blockDim.x + threadIdx.x;
    if (t >= Fpad * GK) return;
    int f = t / GK, k = t % GK;
    Wt[t] = (f < F) ? (__bf16)W[(size_t)k * F + f] : (__bf16)0.0f;
}

// ---------------- WMMA GEMM: C[N,F] = A[N,GK](bf16) x Wt[Fpad,GK]^T (bf16), fp32 acc ----
// One wave -> one 16x16 output tile. 8 waves / block. K-loop fully unrolled (GK=256).
__global__ void gemm_wmma_bf16(const __bf16* __restrict__ A, const __bf16* __restrict__ Bt,
                               float* __restrict__ C, int N, int F, int Fpad) {
    const int lane  = threadIdx.x & 31;
    const int wave  = threadIdx.x >> 5;
    const int mtile = blockIdx.x;
    const int ntile = blockIdx.y * 8 + wave;
    if (ntile * 16 >= Fpad) return;          // wave-uniform: EXEC stays all-ones

    const int m  = lane & 15;                // row (A) / column (B) owned by this lane
    const int kh = lane >> 4;                // K half selector per ISA A/B layout

    int arowi = mtile * 16 + m; if (arowi >= N) arowi = N - 1;   // clamp keeps EXEC full
    const __bf16* arow = A  + (size_t)arowi * GK;
    const __bf16* brow = Bt + (size_t)(ntile * 16 + m) * GK;

    v8f acc = {};
#pragma unroll
    for (int k0 = 0; k0 < GK; k0 += 32) {
        // A lane layout (16-bit A 16x32): VGPR0-3 = K[kh*8 .. kh*8+7], VGPR4-7 = K[16+kh*8 ..]
        v8bf alo = *(const v8bf*)(arow + k0 + kh * 8);
        v8bf ahi = *(const v8bf*)(arow + k0 + 16 + kh * 8);
        v8bf blo = *(const v8bf*)(brow + k0 + kh * 8);
        v8bf bhi = *(const v8bf*)(brow + k0 + 16 + kh * 8);
        v16bf a, b;
#pragma unroll
        for (int i = 0; i < 8; ++i) {
            a[i] = alo[i]; a[8 + i] = ahi[i];
            b[i] = blo[i]; b[8 + i] = bhi[i];
        }
        acc = __builtin_amdgcn_wmma_f32_16x16x32_bf16(
                  false, a, false, b, (short)0, acc, false, false);
    }

    // D layout: VGPR r, lanes 0-15 -> row kh*8+r? (kh=0: rows 0..7; kh=1 half: rows 8..15)
    const int col = ntile * 16 + (lane & 15);
    if (col < F) {
#pragma unroll
        for (int r = 0; r < 8; ++r) {
            int row = mtile * 16 + kh * 8 + r;
            if (row < N) C[(size_t)row * F + col] = acc[r];
        }
    }
}

// ---------------- edge scatter-add: agg[dst] += norm * h[src], float4 vectorized ---------
__global__ void scatter_add4(const float4* __restrict__ h4, const int* __restrict__ src,
                             const int* __restrict__ dst, const float* __restrict__ nrm,
                             float* __restrict__ agg, int E, int F4) {
    long long t = (long long)blockIdx.x * blockDim.x + threadIdx.x;
    int e  = (int)(t / F4);
    int f4 = (int)(t % F4);
    if (e >= E) return;
    float nw = nrm[e];
    float4 v = h4[(size_t)src[e] * F4 + f4];
    float* out = agg + ((size_t)dst[e] * F4 + f4) * 4;
    atomicAdd(out + 0, nw * v.x);
    atomicAdd(out + 1, nw * v.y);
    atomicAdd(out + 2, nw * v.z);
    atomicAdd(out + 3, nw * v.w);
}

// ---------------- epilogue: += self-loop + bias, ELU, bf16 requant for next layer -------
__global__ void epilogue(float* __restrict__ agg, const float* __restrict__ hg,
                         const float* __restrict__ dinv, const float* __restrict__ bias,
                         __bf16* __restrict__ abuf, int N, int F, int do_elu, int wr_bf16) {
    size_t t = (size_t)blockIdx.x * blockDim.x + threadIdx.x;
    if (t >= (size_t)N * F) return;
    int i = (int)(t / F), f = (int)(t % F);
    float di = dinv[i];
    float v = agg[t] + di * di * hg[t] + bias[f];
    if (do_elu) v = v > 0.f ? v : (expf(v) - 1.f);
    agg[t] = v;
    if (wr_bf16) abuf[t] = (__bf16)v;
}

// ---------------- log-softmax over C classes, one wave per row (wave32) ------------------
__global__ void log_softmax_kernel(const float* __restrict__ logits, float* __restrict__ out,
                                   int N, int C) {
    int lane = threadIdx.x & 31;
    int wave = threadIdx.x >> 5;
    int row  = blockIdx.x * 8 + wave;
    if (row >= N) return;
    const float* r = logits + (size_t)row * C;
    float v0 = (lane      < C) ? r[lane]      : -INFINITY;
    float v1 = (lane + 32 < C) ? r[lane + 32] : -INFINITY;
    float m = fmaxf(v0, v1);
    for (int o = 16; o > 0; o >>= 1) m = fmaxf(m, __shfl_xor(m, o, 32));
    float s = ((lane < C) ? expf(v0 - m) : 0.f) + ((lane + 32 < C) ? expf(v1 - m) : 0.f);
    for (int o = 16; o > 0; o >>= 1) s += __shfl_xor(s, o, 32);
    float l = logf(s);
    if (lane      < C) out[(size_t)row * C + lane]      = v0 - m - l;
    if (lane + 32 < C) out[(size_t)row * C + lane + 32] = v1 - m - l;
}

// ---------------- driver ----------------

extern "C" void kernel_launch(void* const* d_in, const int* in_sizes, int n_in,
                              void* d_out, int out_size, void* d_ws, size_t ws_size,
                              hipStream_t stream) {
    const float* x  = (const float*)d_in[0];
    const int*   ei = (const int*)  d_in[1];
    const float* ew = (const float*)d_in[2];
    const float* W1 = (const float*)d_in[3];
    const float* b1 = (const float*)d_in[4];
    const float* W2 = (const float*)d_in[5];
    const float* b2 = (const float*)d_in[6];
    const float* W3 = (const float*)d_in[7];
    const float* b3 = (const float*)d_in[8];

    const int E    = in_sizes[2];          // 800000
    const int N    = in_sizes[0] / GK;     // 50000
    const int H    = GK;                   // 256
    const int C    = in_sizes[8];          // 40
    const int Cpad = (C + 15) & ~15;       // 48
    const int* src = ei;
    const int* dst = ei + E;

    char* ws = (char*)d_ws; size_t off = 0; (void)ws_size;
    auto alloc = [&](size_t bytes) -> void* {
        void* p = ws + off; off += (bytes + 255) & ~(size_t)255; return p;
    };
    __bf16* abuf = (__bf16*)alloc((size_t)N * H * 2);       // activations (bf16 A matrix)
    float*  hg   = (float*) alloc((size_t)N * H * 4);       // GEMM output (pre-aggregate)
    float*  agg  = (float*) alloc((size_t)N * H * 4);       // aggregate / layer output
    float*  deg  = (float*) alloc((size_t)N * 4);
    float*  dinv = (float*) alloc((size_t)N * 4);
    float*  nrm  = (float*) alloc((size_t)E * 4);
    __bf16* Wt1  = (__bf16*)alloc((size_t)H * GK * 2);
    __bf16* Wt2  = (__bf16*)alloc((size_t)H * GK * 2);
    __bf16* Wt3  = (__bf16*)alloc((size_t)Cpad * GK * 2);

    const int T = 256;
    #define GRID1(n) ((unsigned)(((long long)(n) + T - 1) / T))

    // --- normalization coefficients ---
    fill_f32   <<<GRID1(N), T, 0, stream>>>(deg, 1.0f, (size_t)N);   // self-loop weight
    deg_accum  <<<GRID1(E), T, 0, stream>>>(dst, ew, deg, E);
    dinv_kernel<<<GRID1(N), T, 0, stream>>>(deg, dinv, N);
    norm_kernel<<<GRID1(E), T, 0, stream>>>(src, dst, ew, dinv, nrm, E);

    // --- weights -> transposed, padded bf16; x -> bf16 ---
    cvt_wt  <<<GRID1((size_t)H * GK),    T, 0, stream>>>(W1, Wt1, H, H);
    cvt_wt  <<<GRID1((size_t)H * GK),    T, 0, stream>>>(W2, Wt2, H, H);
    cvt_wt  <<<GRID1((size_t)Cpad * GK), T, 0, stream>>>(W3, Wt3, C, Cpad);
    cvt_bf16<<<GRID1((size_t)N * H),     T, 0, stream>>>(x, abuf, (size_t)N * H);

    dim3 gblock(256);

    // --- layer 1 (F=256, ELU) ---
    {
        dim3 grid((unsigned)((N + 15) / 16), (unsigned)((H / 16 + 7) / 8));
        gemm_wmma_bf16<<<grid, gblock, 0, stream>>>(abuf, Wt1, hg, N, H, H);
        fill_f32<<<GRID1((size_t)N * H), T, 0, stream>>>(agg, 0.f, (size_t)N * H);
        long long tot = (long long)E * (H / 4);
        scatter_add4<<<GRID1(tot), T, 0, stream>>>((const float4*)hg, src, dst, nrm, agg, E, H / 4);
        epilogue<<<GRID1((size_t)N * H), T, 0, stream>>>(agg, hg, dinv, b1, abuf, N, H, 1, 1);
    }
    // --- layer 2 (F=256, ELU) ---
    {
        dim3 grid((unsigned)((N + 15) / 16), (unsigned)((H / 16 + 7) / 8));
        gemm_wmma_bf16<<<grid, gblock, 0, stream>>>(abuf, Wt2, hg, N, H, H);
        fill_f32<<<GRID1((size_t)N * H), T, 0, stream>>>(agg, 0.f, (size_t)N * H);
        long long tot = (long long)E * (H / 4);
        scatter_add4<<<GRID1(tot), T, 0, stream>>>((const float4*)hg, src, dst, nrm, agg, E, H / 4);
        epilogue<<<GRID1((size_t)N * H), T, 0, stream>>>(agg, hg, dinv, b2, abuf, N, H, 1, 1);
    }
    // --- layer 3 (F=40, no activation) + log-softmax ---
    {
        dim3 grid((unsigned)((N + 15) / 16), (unsigned)((Cpad / 16 + 7) / 8));
        gemm_wmma_bf16<<<grid, gblock, 0, stream>>>(abuf, Wt3, hg, N, C, Cpad);
        fill_f32<<<GRID1((size_t)N * C), T, 0, stream>>>(agg, 0.f, (size_t)N * C);
        long long tot = (long long)E * (C / 4);
        scatter_add4<<<GRID1(tot), T, 0, stream>>>((const float4*)hg, src, dst, nrm, agg, E, C / 4);
        epilogue<<<GRID1((size_t)N * C), T, 0, stream>>>(agg, hg, dinv, b3, nullptr, N, C, 0, 0);
        log_softmax_kernel<<<(unsigned)((N + 7) / 8), 256, 0, stream>>>(agg, (float*)d_out, N, C);
    }
    #undef GRID1
}
